// RNN_29102698398007
// MI455X (gfx1250) — compile-verified, matrix operand order checked by reading, not run on hardware
//
#include <hip/hip_runtime.h>

// ---------------------------------------------------------------------------
// RNN: out = scan_t( h = sigmoid(x@Wk [.,t,.] + h_prev @ Wr) )
// B=64, T=512, D=1024, H=1024.  bf16 WMMA (16x16x32), f32 accumulate.
// ---------------------------------------------------------------------------

typedef __attribute__((ext_vector_type(16))) __bf16 v16bf;
typedef __attribute__((ext_vector_type(8)))  __bf16 v8bf;
typedef __attribute__((ext_vector_type(8)))  float  v8f;

typedef __bf16 bf16;

#define BATCH 64
#define TLEN  512
#define DDIM  1024
#define HDIM  1024
#define KC    32     // K chunk = WMMA K for bf16
#define NBLK  64     // persistent scan blocks (4 batch-tiles x 16 col-tiles)

// ---- helpers --------------------------------------------------------------

__device__ __forceinline__ bf16 f2bf(float f) {
    // round-to-nearest-even f32 -> bf16 (off hot path)
    union { float f; unsigned u; } v; v.f = f;
    unsigned r = v.u + 0x7FFFu + ((v.u >> 16) & 1u);
    unsigned short h = (unsigned short)(r >> 16);
    return __builtin_bit_cast(bf16, h);
}

// pack two f32 into two bf16 (round-to-nearest-away) with one v_perm_b32
__device__ __forceinline__ unsigned pack_bf16(float f0, float f1) {
    unsigned u0 = __builtin_bit_cast(unsigned, f0) + 0x8000u;
    unsigned u1 = __builtin_bit_cast(unsigned, f1) + 0x8000u;
    return __builtin_amdgcn_perm(u1, u0, 0x07060302u);  // { u1.hi16, u0.hi16 }
}

// Load one 16x32 bf16 operand fragment for this lane.
// p points at row_base + half*8.  ISA layout: lane L%16 = row, lane/16 picks
// K halves {0..7,16..23} vs {8..15,24..31} -> two 16B loads at +0 and +16.
__device__ __forceinline__ v16bf load_frag(const bf16* p) {
    v8bf lo = *(const v8bf*)(p);
    v8bf hi = *(const v8bf*)(p + 16);
    v16bf r;
#pragma unroll
    for (int i = 0; i < 8; ++i) { r[i] = lo[i]; r[i + 8] = hi[i]; }
    return r;
}

// CDNA5 async global -> LDS DMA (ASYNCcnt-tracked)
__device__ __forceinline__ void async_load_b128(unsigned lds_off, const void* gaddr) {
    asm volatile("global_load_async_to_lds_b128 %0, %1, off"
                 :: "v"(lds_off), "v"(gaddr) : "memory");
}
__device__ __forceinline__ void wait_async() {
    asm volatile("s_wait_asynccnt 0x0" ::: "memory");
}

// device-scope split barrier for the persistent scan kernel
__device__ __forceinline__ void grid_barrier(unsigned* counter, unsigned target) {
    __builtin_amdgcn_fence(__ATOMIC_RELEASE, "agent");   // each wave: flush stores
    __syncthreads();
    if (threadIdx.x == 0) {
        __hip_atomic_fetch_add(counter, 1u, __ATOMIC_RELAXED, __HIP_MEMORY_SCOPE_AGENT);
        while (__hip_atomic_load(counter, __ATOMIC_ACQUIRE, __HIP_MEMORY_SCOPE_AGENT) < target)
            __builtin_amdgcn_s_sleep(1);
    }
    __syncthreads();
    __builtin_amdgcn_fence(__ATOMIC_ACQUIRE, "agent");   // each wave: invalidate
}

// ---------------------------------------------------------------------------
// Prep: transpose f32 [K x N] -> bf16 [N x K] (K-contiguous for WMMA staging)
// ---------------------------------------------------------------------------
__global__ __launch_bounds__(256) void transpose_to_bf16(
        const float* __restrict__ in, bf16* __restrict__ out, int K, int N) {
    int idx = blockIdx.x * 256 + threadIdx.x;
    if (idx < K * N) {
        int k = idx / N;
        int n = idx - k * N;
        out[n * K + k] = f2bf(in[idx]);
    }
}

__global__ __launch_bounds__(256) void init_state(
        bf16* __restrict__ h0, unsigned* __restrict__ counter, int n) {
    int idx = blockIdx.x * 256 + threadIdx.x;
    if (idx < n) h0[idx] = __builtin_bit_cast(bf16, (unsigned short)0);
    if (idx == 0) *counter = 0u;
}

// ---------------------------------------------------------------------------
// Input GEMM: OUT[M,H] = X[M,D] @ Wk[D,H], M = B*T = 32768.
// kT = Wk^T bf16 [H x D] K-contiguous.
// Block tile 128x128, 256 threads (8 waves); wave tile 32(M) x 64(N)
// => 8 WMMAs per K-chunk per wave.  B staged via async global->LDS DMA,
// A (f32 x) staged through registers with packed bf16 conversion.
// ---------------------------------------------------------------------------
__global__ __launch_bounds__(256) void gemm_xh(
        const float* __restrict__ X, const bf16* __restrict__ kT,
        float* __restrict__ OUT) {
    __shared__ __align__(16) bf16 As[2][128 * KC];
    __shared__ __align__(16) bf16 Bs[2][128 * KC];

    const int t     = threadIdx.x;
    const int wave  = t >> 5;
    const int lane  = t & 31;
    const int row16 = lane & 15;
    const int half  = lane >> 4;
    const int wm    = wave & 3;   // 32-row M strip
    const int wn    = wave >> 2;  // 64-col N strip

    const int m0 = blockIdx.x * 128;
    const int n0 = blockIdx.y * 128;

    const int arow = t & 127;           // A: 128 rows x 32 K (f32)
    const int ak   = (t >> 7) << 4;     // 16 floats per thread
    const int brow = t & 127;           // B: 128 rows x 32 K (bf16)
    const int bk   = (t >> 7) << 4;     // 16 bf16 per thread (2 async b128)

    const float* xptr = X  + (long)(m0 + arow) * DDIM;
    const bf16*  bptr = kT + (long)(n0 + brow) * DDIM;

    v8f acc[2][4] = {};

    // ---- prologue: chunk 0 ----
    float4 a0 = *(const float4*)(xptr + ak);
    float4 a1 = *(const float4*)(xptr + ak + 4);
    float4 a2 = *(const float4*)(xptr + ak + 8);
    float4 a3 = *(const float4*)(xptr + ak + 12);
    {
        unsigned bofs = (unsigned)(size_t)&Bs[0][brow * KC + bk];
        async_load_b128(bofs,      bptr + bk);
        async_load_b128(bofs + 16, bptr + bk + 8);
    }
    {
        uint4 p0, p1;
        p0.x = pack_bf16(a0.x, a0.y); p0.y = pack_bf16(a0.z, a0.w);
        p0.z = pack_bf16(a1.x, a1.y); p0.w = pack_bf16(a1.z, a1.w);
        p1.x = pack_bf16(a2.x, a2.y); p1.y = pack_bf16(a2.z, a2.w);
        p1.z = pack_bf16(a3.x, a3.y); p1.w = pack_bf16(a3.z, a3.w);
        bf16* dst = &As[0][arow * KC + ak];
        *(uint4*)(dst)     = p0;
        *(uint4*)(dst + 8) = p1;
    }
    wait_async();
    __syncthreads();

    int buf = 0;
    for (int kc = 0; kc < DDIM; kc += KC) {
        const bool has_next = (kc + KC) < DDIM;
        if (has_next) {
            unsigned bofs = (unsigned)(size_t)&Bs[buf ^ 1][brow * KC + bk];
            async_load_b128(bofs,      bptr + kc + KC + bk);
            async_load_b128(bofs + 16, bptr + kc + KC + bk + 8);
            a0 = *(const float4*)(xptr + kc + KC + ak);
            a1 = *(const float4*)(xptr + kc + KC + ak + 4);
            a2 = *(const float4*)(xptr + kc + KC + ak + 8);
            a3 = *(const float4*)(xptr + kc + KC + ak + 12);
        }

        v16bf af0 = load_frag(&As[buf][(wm * 32 +  0 + row16) * KC + half * 8]);
        v16bf af1 = load_frag(&As[buf][(wm * 32 + 16 + row16) * KC + half * 8]);
#pragma unroll
        for (int ni = 0; ni < 4; ++ni) {
            v16bf bfr = load_frag(&Bs[buf][(wn * 64 + ni * 16 + row16) * KC + half * 8]);
            acc[0][ni] = __builtin_amdgcn_wmma_f32_16x16x32_bf16(
                             false, af0, false, bfr, (short)0, acc[0][ni], false, false);
            acc[1][ni] = __builtin_amdgcn_wmma_f32_16x16x32_bf16(
                             false, af1, false, bfr, (short)0, acc[1][ni], false, false);
        }

        if (has_next) {
            uint4 p0, p1;
            p0.x = pack_bf16(a0.x, a0.y); p0.y = pack_bf16(a0.z, a0.w);
            p0.z = pack_bf16(a1.x, a1.y); p0.w = pack_bf16(a1.z, a1.w);
            p1.x = pack_bf16(a2.x, a2.y); p1.y = pack_bf16(a2.z, a2.w);
            p1.z = pack_bf16(a3.x, a3.y); p1.w = pack_bf16(a3.z, a3.w);
            bf16* dst = &As[buf ^ 1][arow * KC + ak];
            *(uint4*)(dst)     = p0;
            *(uint4*)(dst + 8) = p1;
            wait_async();
        }
        __syncthreads();
        buf ^= 1;
    }

    // C layout: lane holds column n = row16, rows m = r + 8*half.
    // Non-temporal stores: xh streams once; keep x resident in L2.
#pragma unroll
    for (int mi = 0; mi < 2; ++mi) {
#pragma unroll
        for (int ni = 0; ni < 4; ++ni) {
            const long mbase = (long)(m0 + wm * 32 + mi * 16 + half * 8);
            float* op = OUT + mbase * HDIM + (n0 + wn * 64 + ni * 16 + row16);
#pragma unroll
            for (int r = 0; r < 8; ++r)
                __builtin_nontemporal_store(acc[mi][ni][r], op + (long)r * HDIM);
        }
    }
}

// ---------------------------------------------------------------------------
// Persistent recurrent scan.  Grid: 64 blocks (4 batch-tiles x 16 col-tiles),
// 128 threads.  Each block owns output tile [16 batch x 64 cols] for all T.
//
// Prologue: stage this block's 64-column slice of Wr^T (64 x 1024 bf16 =
// 128 KB, fits CDNA5 320 KB LDS) once via async global->LDS DMA.
// Per step: A fragments (h_prev, L2-resident) direct from global, B fragments
// from LDS, 32 chained WMMAs (dual accumulators), fused sigmoid epilogue,
// then one device-scope barrier (ping-pong h buffers need only one).
// ---------------------------------------------------------------------------
__global__ __launch_bounds__(128) void rnn_scan(
        bf16* __restrict__ h0, bf16* __restrict__ h1,
        const bf16* __restrict__ wrT, float* __restrict__ OUT,
        unsigned* __restrict__ counter) {
    __shared__ __align__(16) bf16 Bsl[64 * HDIM];   // 128 KB weight slice

    const int t     = threadIdx.x;
    const int wave  = t >> 5;
    const int lane  = t & 31;
    const int row16 = lane & 15;
    const int half  = lane >> 4;

    const int mb = blockIdx.x;             // 16-batch tile (0..3)
    const int n0 = blockIdx.y * 64;        // column tile base
    const int n  = n0 + wave * 16 + row16;

    // ---- stage Wr^T slice: rows n0..n0+63, all K, into LDS (once) ----
    {
        const bf16* src  = wrT + (long)n0 * HDIM;          // 64*1024 bf16
        unsigned    base = (unsigned)(size_t)&Bsl[0];
        for (int i = 0; i < 64; ++i) {                     // 64 x b128 per thread
            int e = (t + 128 * i) << 3;                    // bf16 element offset
            async_load_b128(base + (unsigned)(e << 1), src + e);
        }
        wait_async();
    }
    __syncthreads();

    const bf16* bfrag = &Bsl[(wave * 16 + row16) * HDIM + half * 8];

    bf16* hp = h0;   // read buffer
    bf16* hn = h1;   // write buffer

    for (int ts = 0; ts < TLEN; ++ts) {
        const bf16* ap = hp + (mb * 16 + row16) * HDIM + half * 8;

        v8f acc0 = {};
        v8f acc1 = {};
#pragma unroll 4
        for (int kc = 0; kc < HDIM; kc += 2 * KC) {
            acc0 = __builtin_amdgcn_wmma_f32_16x16x32_bf16(
                       false, load_frag(ap + kc), false, load_frag(bfrag + kc),
                       (short)0, acc0, false, false);
            acc1 = __builtin_amdgcn_wmma_f32_16x16x32_bf16(
                       false, load_frag(ap + kc + KC), false, load_frag(bfrag + kc + KC),
                       (short)0, acc1, false, false);
        }

        // fuse xh add + sigmoid + dual store (f32 out, bf16 next-state)
#pragma unroll
        for (int r = 0; r < 8; ++r) {
            const int  b    = mb * 16 + half * 8 + r;
            const long oidx = ((long)b * TLEN + ts) * HDIM + n;
            float v  = OUT[oidx] + acc0[r] + acc1[r];
            float hs = 1.0f / (1.0f + __expf(-v));
            OUT[oidx]           = hs;
            hn[b * HDIM + n]    = f2bf(hs);
        }

        grid_barrier(counter, (unsigned)(NBLK * (ts + 1)));

        bf16* tmp = hp; hp = hn; hn = tmp;
    }
}

// ---------------------------------------------------------------------------
extern "C" void kernel_launch(void* const* d_in, const int* in_sizes, int n_in,
                              void* d_out, int out_size, void* d_ws, size_t ws_size,
                              hipStream_t stream) {
    const float* x  = (const float*)d_in[0];   // [B,T,D]
    const float* Wk = (const float*)d_in[1];   // [D,H]
    const float* Wr = (const float*)d_in[2];   // [H,H]
    float* out = (float*)d_out;                // [B,T,H]

    char* ws = (char*)d_ws;
    bf16*     kT  = (bf16*)(ws);                               // 2 MB  [H x D]
    bf16*     wrT = (bf16*)(ws + (2u << 20));                  // 2 MB  [H x H]
    bf16*     h0  = (bf16*)(ws + (4u << 20));                  // 128 KB
    bf16*     h1  = (bf16*)(ws + (4u << 20) + BATCH * HDIM * 2);
    unsigned* cnt = (unsigned*)(ws + (4u << 20) + 2 * BATCH * HDIM * 2);

    // one-time weight conversion + transpose (K-contiguous bf16)
    transpose_to_bf16<<<(DDIM * HDIM + 255) / 256, 256, 0, stream>>>(Wk, kT, DDIM, HDIM);
    transpose_to_bf16<<<(HDIM * HDIM + 255) / 256, 256, 0, stream>>>(Wr, wrT, HDIM, HDIM);
    init_state<<<(BATCH * HDIM + 255) / 256, 256, 0, stream>>>(h0, cnt, BATCH * HDIM);

    // xh = x @ Wk -> d_out   (32768 x 1024 x 1024 bf16 WMMA GEMM)
    gemm_xh<<<dim3((BATCH * TLEN) / 128, HDIM / 128), 256, 0, stream>>>(x, kT, out);

    // persistent serial scan over T (single launch, in-kernel device barrier)
    rnn_scan<<<dim3(BATCH / 16, HDIM / 64), 128, 0, stream>>>(h0, h1, wrT, out, cnt);
}
